// MTNN_35098472742913
// MI455X (gfx1250) — matrix operand order repaired and checked by reading, not compiled, not run on hardware
//
#include <hip/hip_runtime.h>
#include <hip/hip_bf16.h>

typedef __attribute__((ext_vector_type(16))) __bf16 v16bf;
typedef __attribute__((ext_vector_type(8)))  __bf16 v8bf;
typedef __attribute__((ext_vector_type(4)))  __bf16 v4bf;
typedef __attribute__((ext_vector_type(8)))  float  v8f;
typedef __attribute__((ext_vector_type(4)))  int    v4i;

#define BB 512
#define TT 200
#define HS 256
#define HD 128
#define DD 512   // 2*HD + HS
#define LDP 136  // padded LDS stride in halves: 272B = 16B-aligned, 4-bank rotate

#if defined(__gfx1250__) && __has_builtin(__builtin_amdgcn_global_load_async_to_lds_b128)
#define HAVE_ASYNC_LDS 1
#endif

template <typename T>
__device__ __forceinline__ __bf16 to_bf(T x) { return (__bf16)(float)x; }

// 8 contiguous elements -> bf16 halves a[o..o+7]
__device__ __forceinline__ void ld8(const float* __restrict__ p, v16bf& a, int o) {
  float4 f0 = *(const float4*)(p);
  float4 f1 = *(const float4*)(p + 4);
  a[o + 0] = (__bf16)f0.x; a[o + 1] = (__bf16)f0.y;
  a[o + 2] = (__bf16)f0.z; a[o + 3] = (__bf16)f0.w;
  a[o + 4] = (__bf16)f1.x; a[o + 5] = (__bf16)f1.y;
  a[o + 6] = (__bf16)f1.z; a[o + 7] = (__bf16)f1.w;
}
__device__ __forceinline__ void ld8(const __bf16* __restrict__ p, v16bf& a, int o) {
  v8bf v = *(const v8bf*)(p);
#pragma unroll
  for (int i = 0; i < 8; ++i) a[o + i] = v[i];
}

// ---------------------------------------------------------------------------
// Generic GEMM: C[M,N] = act(A[M,K] @ W[N,K]^T + bias1 + bias2), C in bf16.
// One wave per 16x16 tile; full K-chunks use b128 loads, tail is predicated.
// ---------------------------------------------------------------------------
template <typename TA>
__global__ void gemm_bias_kernel(const TA* __restrict__ A, int lda,
                                 const float* __restrict__ W, int ldw,
                                 const float* __restrict__ bias1,
                                 const float* __restrict__ bias2,
                                 __bf16* __restrict__ C, int ldc,
                                 int M, int N, int K, int do_relu) {
  const int lane = threadIdx.x & 31;
  const int wid  = blockIdx.x * (blockDim.x >> 5) + (threadIdx.x >> 5);
  const int ntn  = N >> 4;
  const int tilesTotal = (M >> 4) * ntn;
  if (wid >= tilesTotal) return;            // whole wave exits together
  const int tm = wid / ntn, tn = wid % ntn;

  const int m   = lane & 15;
  const int oct = (lane >> 4) * 8;          // unused; A loads also use kh
  const int n   = lane & 15;
  const int kh  = (lane >> 4) * 16;         // B K-half base

  const TA*    Arow = A + (size_t)(tm * 16 + m) * lda;
  const float* Wrow = W + (size_t)(tn * 16 + n) * ldw;

  v8f acc = {0.f, 0.f, 0.f, 0.f, 0.f, 0.f, 0.f, 0.f};

  const int kfull = K & ~31;
  int kb = 0;
  for (; kb < kfull; kb += 32) {
    v16bf a, b;
    ld8(Arow + kb + kh, a, 0);
    ld8(Arow + kb + kh + 8, a, 8);
    ld8(Wrow + kb + kh, b, 0);
    ld8(Wrow + kb + kh + 8, b, 8);
    acc = __builtin_amdgcn_wmma_f32_16x16x32_bf16(
        false, a, false, b, (short)0, acc, false, false);
  }
  if (kb < K) {                             // predicated tail chunk
    v16bf a, b;
#pragma unroll
    for (int i = 0; i < 8; ++i) {
      int k0 = kb + kh + i;
      int k1 = kb + kh + 8 + i;
      a[i]     = (k0 < K) ? to_bf(Arow[k0]) : (__bf16)0.0f;
      a[8 + i] = (k1 < K) ? to_bf(Arow[k1]) : (__bf16)0.0f;
    }
#pragma unroll
    for (int i = 0; i < 16; ++i) {
      int k = kb + kh + i;
      b[i] = (k < K) ? (__bf16)Wrow[k] : (__bf16)0.0f;
    }
    acc = __builtin_amdgcn_wmma_f32_16x16x32_bf16(
        false, a, false, b, (short)0, acc, false, false);
  }

  const int colN = tn * 16 + (lane & 15);
  float bv = bias1 ? bias1[colN] : 0.f;
  if (bias2) bv += bias2[colN];
  size_t ci = (size_t)(tm * 16 + 8 * (lane >> 4)) * ldc + colN;
#pragma unroll
  for (int r = 0; r < 8; ++r) {
    float v = acc[r] + bv;
    if (do_relu) v = fmaxf(v, 0.f);
    C[ci] = (__bf16)v;
    ci += ldc;
  }
}

// ---------------------------------------------------------------------------
// Bidirectional relu-RNN recurrence, persistent time loop.
// gridDim = (B/16, 2): block owns 16 batch rows; blockIdx.y = direction.
// 8 waves/block; wave wv owns output columns [wv*16, wv*16+16).
// W_hh B-fragments hoisted in registers for all T steps.
// xw tile streams in via async LDS-DMA (b128/thread) overlapped with the
// WMMA chain; hidden state redistributes C-layout -> A-layout through LDS
// with b128 LDS loads (each A half-fragment is 8 contiguous aligned halves).
// ---------------------------------------------------------------------------
__global__ void rnn_kernel(const __bf16* __restrict__ xw_f,
                           const __bf16* __restrict__ xw_b,
                           const float* __restrict__ whh_f,
                           const float* __restrict__ whh_b,
                           __bf16* __restrict__ out_f,
                           __bf16* __restrict__ out_b,
                           int T) {
  __shared__ __bf16 sXW[16 * LDP];
  __shared__ __bf16 sH[16 * LDP];

  const int dir  = blockIdx.y;
  const int b0   = blockIdx.x * 16;
  const int tid  = threadIdx.x;
  const int lane = tid & 31;
  const int wv   = tid >> 5;                // ntile = wave id (0..7)

  const __bf16* xw  = dir ? xw_b  : xw_f;
  const float*  whh = dir ? whh_b : whh_f;
  __bf16*       out = dir ? out_b : out_f;

  const int n   = lane & 15;
  const int kh  = (lane >> 4) * 16;         // B layout
  const int m   = lane & 15;
  const int oct = (lane >> 4) * 8;          // unused; A layout also uses kh

  // cooperative tile mover: thread -> (row, 8-half chunk)
  const int crow = tid >> 4;                // 0..15
  const int cch  = tid & 15;                // 0..15
  const int t0   = dir ? (T - 1) : 0;
  const ptrdiff_t dstep = (ptrdiff_t)(dir ? -HD : HD);
  const __bf16* xwp  = xw  + ((size_t)(b0 + crow) * T + t0) * HD + cch * 8;
  __bf16*       outp = out + ((size_t)(b0 + crow) * T + t0) * HD + cch * 8;
  __bf16* sXWp = &sXW[crow * LDP + cch * 8];
  const __bf16* sHp = &sH[crow * LDP + cch * 8];

  // Hoist B = W_hh^T fragments: B[k][j] = whh[j*HD + k], j = wv*16+n.
  v16bf Bf[4];
  {
    const float* wrow = whh + (size_t)(wv * 16 + n) * HD;
#pragma unroll
    for (int kc = 0; kc < 4; ++kc) {
      ld8(wrow + kc * 32 + kh, Bf[kc], 0);
      ld8(wrow + kc * 32 + kh + 8, Bf[kc], 8);
    }
  }

  // h0 = 0
  v16bf hA[4];
#pragma unroll
  for (int kc = 0; kc < 4; ++kc)
#pragma unroll
    for (int i = 0; i < 16; ++i) hA[kc][i] = (__bf16)0.0f;

  for (int s = 0; s < T; ++s) {
    // ---- stream in xw tile for this step (overlap with WMMA chain) ----
#if defined(HAVE_ASYNC_LDS)
    __builtin_amdgcn_global_load_async_to_lds_b128(
        (v4i*)xwp,
        (__attribute__((address_space(3))) v4i*)sXWp, 0, 0);
#else
    v8bf xv = *(const v8bf*)xwp;
#endif
    xwp += dstep;

    v8f acc = {0.f, 0.f, 0.f, 0.f, 0.f, 0.f, 0.f, 0.f};
#pragma unroll
    for (int kc = 0; kc < 4; ++kc)
      acc = __builtin_amdgcn_wmma_f32_16x16x32_bf16(
          false, hA[kc], false, Bf[kc], (short)0, acc, false, false);

#if defined(HAVE_ASYNC_LDS)
#if __has_builtin(__builtin_amdgcn_s_wait_asynccnt)
    __builtin_amdgcn_s_wait_asynccnt(0);
#else
    asm volatile("s_wait_asynccnt 0" ::: "memory");
#endif
#else
    *(v8bf*)sXWp = xv;
#endif
    __syncthreads();                        // sXW tile complete

    // h_new = relu(h @ W^T + xw); write to sH in C layout
#pragma unroll
    for (int r = 0; r < 8; ++r) {
      int row = r + 8 * (lane >> 4);        // 0..15 within tile
      float v = acc[r] + (float)sXW[row * LDP + wv * 16 + n];
      v = fmaxf(v, 0.f);
      sH[row * LDP + wv * 16 + n] = (__bf16)v;
    }
    __syncthreads();                        // sH tile complete

    // cooperative coalesced store of h row to global (b128 per thread)
    *(v8bf*)outp = *(const v8bf*)sHp;
    outp += dstep;

    // regather A fragments for next step: two b128 LDS loads per K-chunk
#pragma unroll
    for (int kc = 0; kc < 4; ++kc) {
      const __bf16* hr = sH + m * LDP + kc * 32;
      v8bf lo = *(const v8bf*)(hr + kh);
      v8bf hi = *(const v8bf*)(hr + kh + 8);
#pragma unroll
      for (int i = 0; i < 8; ++i) {
        hA[kc][i]     = lo[i];
        hA[kc][8 + i] = hi[i];
      }
    }
    __syncthreads();                        // sH consumed
  }
}

// ---------------------------------------------------------------------------
// Routed per-sample dot: out[b,t] = relu(<[s|fwd|bwd], Wn[order[b]]> + bias).
// One block per batch element; weight row cached in LDS; static-branch dot
// is time-invariant. All global reads are b64/b128 per lane.
// ---------------------------------------------------------------------------
__global__ void route_dot_kernel(const __bf16* __restrict__ s,
                                 const __bf16* __restrict__ fwd,
                                 const __bf16* __restrict__ bwd,
                                 const int* __restrict__ order,
                                 const float* __restrict__ nw,
                                 const float* __restrict__ nb,
                                 float* __restrict__ out, int T) {
  __shared__ float sWn[DD];
  const int b    = blockIdx.x;
  const int tid  = threadIdx.x;
  const int lane = tid & 31;
  const int wv   = tid >> 5;                // 8 waves

  const int nn = order[b];
  for (int i = tid; i < DD; i += blockDim.x) sWn[i] = nw[(size_t)nn * DD + i];
  __syncthreads();
  const float bias = nb[nn];

  // time-invariant static contribution (per-wave copy), b128 per lane
  float sd = 0.f;
  {
    v8bf sv = *(const v8bf*)(s + (size_t)b * HS + lane * 8);
#pragma unroll
    for (int j = 0; j < 8; ++j) sd += (float)sv[j] * sWn[lane * 8 + j];
#pragma unroll
    for (int o = 16; o; o >>= 1) sd += __shfl_xor(sd, o, 32);
  }

  for (int t = wv; t < T; t += 8) {
    float p = 0.f;
    v4bf fv = *(const v4bf*)(fwd + ((size_t)b * T + t) * HD + lane * 4);
    v4bf bv = *(const v4bf*)(bwd + ((size_t)b * T + t) * HD + lane * 4);
#pragma unroll
    for (int j = 0; j < 4; ++j) {
      p += (float)fv[j] * sWn[HS + lane * 4 + j];
      p += (float)bv[j] * sWn[HS + HD + lane * 4 + j];
    }
#pragma unroll
    for (int o = 16; o; o >>= 1) p += __shfl_xor(p, o, 32);
    if (lane == 0) out[(size_t)b * T + t] = fmaxf(p + sd + bias, 0.f);
  }
}

// ---------------------------------------------------------------------------
extern "C" void kernel_launch(void* const* d_in, const int* in_sizes, int n_in,
                              void* d_out, int out_size, void* d_ws, size_t ws_size,
                              hipStream_t stream) {
  (void)in_sizes; (void)n_in; (void)out_size; (void)ws_size;

  const float* x_static  = (const float*)d_in[0];
  const float* x_dynamic = (const float*)d_in[1];
  const int*   order     = (const int*)  d_in[2];
  const float* w_s1 = (const float*)d_in[3];  const float* b_s1 = (const float*)d_in[4];
  const float* w_s2 = (const float*)d_in[5];  const float* b_s2 = (const float*)d_in[6];
  const float* w_dyn = (const float*)d_in[7]; const float* b_dyn = (const float*)d_in[8];
  const float* w_ih_f = (const float*)d_in[9];  const float* w_hh_f = (const float*)d_in[10];
  const float* b_ih_f = (const float*)d_in[11]; const float* b_hh_f = (const float*)d_in[12];
  const float* w_ih_b = (const float*)d_in[13]; const float* w_hh_b = (const float*)d_in[14];
  const float* b_ih_b = (const float*)d_in[15]; const float* b_hh_b = (const float*)d_in[16];
  const float* nweights = (const float*)d_in[17];
  const float* nbias    = (const float*)d_in[18];
  float* out = (float*)d_out;

  // workspace carve (bf16 intermediates), 256B aligned
  char* wp = (char*)d_ws;
  auto carve = [&](size_t bytes) {
    void* p = (void*)wp;
    wp += (bytes + 255) & ~(size_t)255;
    return p;
  };
  __bf16* s1  = (__bf16*)carve((size_t)BB * HS * 2);
  __bf16* s2  = (__bf16*)carve((size_t)BB * HS * 2);
  __bf16* dbf = (__bf16*)carve((size_t)BB * TT * HD * 2);
  __bf16* xwf = (__bf16*)carve((size_t)BB * TT * HD * 2);
  __bf16* xwb = (__bf16*)carve((size_t)BB * TT * HD * 2);
  __bf16* ffw = (__bf16*)carve((size_t)BB * TT * HD * 2);
  __bf16* fbw = (__bf16*)carve((size_t)BB * TT * HD * 2);

  const int BT = BB * TT;

  // 1. static layer 1: [512,100] x [100,256]
  {
    int tiles = (BB / 16) * (HS / 16), blocks = (tiles + 3) / 4;
    gemm_bias_kernel<float><<<blocks, 128, 0, stream>>>(
        x_static, 100, w_s1, 100, b_s1, nullptr, s1, HS, BB, HS, 100, 1);
  }
  // 2. static layer 2: [512,256] x [256,256]
  {
    int tiles = (BB / 16) * (HS / 16), blocks = (tiles + 3) / 4;
    gemm_bias_kernel<__bf16><<<blocks, 128, 0, stream>>>(
        s1, HS, w_s2, HS, b_s2, nullptr, s2, HS, BB, HS, HS, 1);
  }
  // 3. dynamic projection: [102400,68] x [68,128]
  {
    int tiles = (BT / 16) * (HD / 16), blocks = (tiles + 3) / 4;
    gemm_bias_kernel<float><<<blocks, 128, 0, stream>>>(
        x_dynamic, 68, w_dyn, 68, b_dyn, nullptr, dbf, HD, BT, HD, 68, 1);
  }
  // 4./5. RNN input projections (bias = b_ih + b_hh, no relu)
  {
    int tiles = (BT / 16) * (HD / 16), blocks = (tiles + 3) / 4;
    gemm_bias_kernel<__bf16><<<blocks, 128, 0, stream>>>(
        dbf, HD, w_ih_f, HD, b_ih_f, b_hh_f, xwf, HD, BT, HD, HD, 0);
    gemm_bias_kernel<__bf16><<<blocks, 128, 0, stream>>>(
        dbf, HD, w_ih_b, HD, b_ih_b, b_hh_b, xwb, HD, BT, HD, HD, 0);
  }
  // 6. bidirectional recurrence (persistent time loop, W_hh in registers)
  rnn_kernel<<<dim3(BB / 16, 2), 256, 0, stream>>>(
      xwf, xwb, w_hh_f, w_hh_b, ffw, fbw, TT);
  // 7. routed dot + bias + relu
  route_dot_kernel<<<BB, 256, 0, stream>>>(
      s2, ffw, fbw, order, nweights, nbias, out, TT);
}
